// ConfidenceAdaptiveSystem_70703751627392
// MI455X (gfx1250) — compile-verified, matrix operand order checked by reading, not run on hardware
//
#include <hip/hip_runtime.h>
#include <hip/hip_bf16.h>

// CDNA5 / gfx1250 fused confidence-adaptive MoE:
//   layer1 GEMMs via v_wmma_f32_16x16x32_bf16 (f32 accumulate),
//   layer2 (H -> C=2) fused into the GEMM epilogue via shfl+LDS+global atomics,
//   final 2-class softmax-confidence select in a tiny second kernel.

typedef __attribute__((ext_vector_type(16))) __bf16 v16bf;
typedef __attribute__((ext_vector_type(8)))  __bf16 v8bf;
typedef __attribute__((ext_vector_type(8)))  float  v8f;

#define BSZ 8192
#define DD  4096
#define HH  8192
#define BM  128
#define BN  128
#define BK  32
#define LDA 40   // LDS row stride in bf16 elements (80B: 16B-aligned, bank-spread)

// f32 -> bf16 round-to-nearest-even, packed pair into one dword
__device__ __forceinline__ unsigned pkbf(float a, float b) {
    unsigned ua = __float_as_uint(a); ua += 0x7FFFu + ((ua >> 16) & 1u);
    unsigned ub = __float_as_uint(b); ub += 0x7FFFu + ((ub >> 16) & 1u);
    return (ua >> 16) | (ub & 0xFFFF0000u);
}

__global__ __launch_bounds__(256) void fused_expert_gemm(
    const float* __restrict__ x,
    const float* __restrict__ t_w1, const float* __restrict__ t_b1, const float* __restrict__ t_w2,
    const float* __restrict__ f_w1, const float* __restrict__ f_b1, const float* __restrict__ f_w2,
    float* __restrict__ acc)   // [2][BSZ][2] f32, pre-zeroed
{
    const int expert = blockIdx.z;
    const float* __restrict__ w1 = expert ? f_w1 : t_w1;
    const float* __restrict__ b1 = expert ? f_b1 : t_b1;
    const float* __restrict__ w2 = expert ? f_w2 : t_w2;
    float* __restrict__ accE = acc + (size_t)expert * (BSZ * 2);

    __shared__ unsigned short xlds[BM * LDA];   // A tile, bf16, [m][k]
    __shared__ unsigned short wlds[BN * LDA];   // B tile, bf16, transposed [n][k]
    __shared__ float ldsAcc[BM * 2];            // per-block layer2 partials

    const int tid   = threadIdx.x;
    const int lane  = tid & 31;
    const int wave  = tid >> 5;
    const int waveM = wave & 3;   // 0..3 -> 32-row slab
    const int waveN = wave >> 2;  // 0..1 -> 64-col slab
    const int rowBase = blockIdx.y * BM;
    const int colBase = blockIdx.x * BN;

    ldsAcc[tid] = 0.0f;           // BM*2 == 256 == blockDim.x

    v8f c[2][4];
    v8f vz = {};
#pragma unroll
    for (int sm = 0; sm < 2; ++sm)
#pragma unroll
        for (int sn = 0; sn < 4; ++sn) c[sm][sn] = vz;

    const int khalf = (lane & 16) ? 8 : 0;   // 16-bit A/B layout: half-K select
    const int rsel  = lane & 15;

    // global->LDS staging patterns
    const int xm  = tid >> 3;         // +p*32 : row within x tile
    const int xk  = (tid & 7) * 4;    // float4 column within 32-wide K slab
    const int wn  = tid & 127;        // column of w1 tile (row of transposed LDS tile)
    const int wkq = (tid >> 7) * 4;   // +p*8 : k-quad base

    for (int kt = 0; kt < DD / BK; ++kt) {
        const int k0 = kt * BK;

        float4 xr[4];
        float  wr[4][4];
#pragma unroll
        for (int p = 0; p < 4; ++p) {
            const float* px = x + (size_t)(rowBase + xm + p * 32) * DD + (k0 + xk);
            xr[p] = *(const float4*)px;
#pragma unroll
            for (int i = 0; i < 4; ++i)
                wr[p][i] = w1[(size_t)(k0 + wkq + p * 8 + i) * HH + (colBase + wn)];
        }

        __syncthreads();
#pragma unroll
        for (int p = 0; p < 4; ++p) {
            *(uint2*)&xlds[(xm + p * 32) * LDA + xk] =
                make_uint2(pkbf(xr[p].x, xr[p].y), pkbf(xr[p].z, xr[p].w));
            *(uint2*)&wlds[wn * LDA + wkq + p * 8] =
                make_uint2(pkbf(wr[p][0], wr[p][1]), pkbf(wr[p][2], wr[p][3]));
        }
        __syncthreads();

        v16bf a[2], b[4];
#pragma unroll
        for (int sm = 0; sm < 2; ++sm) {
            const unsigned short* pa = &xlds[(waveM * 32 + sm * 16 + rsel) * LDA + khalf];
            v8bf lo = *(const v8bf*)pa;           // K = khalf + 0..7
            v8bf hi = *(const v8bf*)(pa + 16);    // K = 16 + khalf + 0..7
            a[sm] = __builtin_shufflevector(lo, hi, 0,1,2,3,4,5,6,7,8,9,10,11,12,13,14,15);
        }
#pragma unroll
        for (int sn = 0; sn < 4; ++sn) {
            const unsigned short* pb = &wlds[(waveN * 64 + sn * 16 + rsel) * LDA + khalf];
            v8bf lo = *(const v8bf*)pb;
            v8bf hi = *(const v8bf*)(pb + 16);
            b[sn] = __builtin_shufflevector(lo, hi, 0,1,2,3,4,5,6,7,8,9,10,11,12,13,14,15);
        }

#pragma unroll
        for (int sm = 0; sm < 2; ++sm)
#pragma unroll
            for (int sn = 0; sn < 4; ++sn)
                c[sm][sn] = __builtin_amdgcn_wmma_f32_16x16x32_bf16(
                    false, a[sm], false, b[sn], (short)0, c[sm][sn], false, false);
    }

    // ---- fused layer 2: out[m,cc] += sum_n relu(h[m,n]+b1[n]) * w2[n,cc] ----
    float b1v[4], w20[4], w21[4];
#pragma unroll
    for (int sn = 0; sn < 4; ++sn) {
        const int ng = colBase + waveN * 64 + sn * 16 + rsel;
        b1v[sn] = b1[ng];
        w20[sn] = w2[ng * 2 + 0];
        w21[sn] = w2[ng * 2 + 1];
    }

#pragma unroll
    for (int sm = 0; sm < 2; ++sm) {
#pragma unroll
        for (int v = 0; v < 8; ++v) {
            float r0 = 0.0f, r1 = 0.0f;
#pragma unroll
            for (int sn = 0; sn < 4; ++sn) {
                float h = c[sm][sn][v] + b1v[sn];
                h = h > 0.0f ? h : 0.0f;
                r0 += h * w20[sn];
                r1 += h * w21[sn];
            }
            // reduce over the 16 lanes that share this output row (wave32)
#pragma unroll
            for (int m = 1; m < 16; m <<= 1) {
                r0 += __shfl_xor(r0, m, 32);
                r1 += __shfl_xor(r1, m, 32);
            }
            if (rsel == 0) {
                const int mLocal = waveM * 32 + sm * 16 + v + ((lane & 16) ? 8 : 0);
                atomicAdd(&ldsAcc[mLocal * 2 + 0], r0);   // ds_add_f32
                atomicAdd(&ldsAcc[mLocal * 2 + 1], r1);
            }
        }
    }
    __syncthreads();

    {   // one global atomic per (row, class) per block
        const int m  = tid >> 1;
        const int cc = tid & 1;
        atomicAdd(&accE[(size_t)(rowBase + m) * 2 + cc], ldsAcc[m * 2 + cc]);
    }
}

__global__ void confidence_select(const float* __restrict__ acc,
                                  const float* __restrict__ t_b2,
                                  const float* __restrict__ f_b2,
                                  float* __restrict__ out)
{
    const int r = blockIdx.x * blockDim.x + threadIdx.x;
    if (r >= BSZ) return;
    const float t0 = acc[r * 2 + 0] + t_b2[0];
    const float t1 = acc[r * 2 + 1] + t_b2[1];
    const float f0 = acc[BSZ * 2 + r * 2 + 0] + f_b2[0];
    const float f1 = acc[BSZ * 2 + r * 2 + 1] + f_b2[1];
    // 2-class softmax max-prob = sigmoid(|t0 - t1|)
    const float conf = 1.0f / (1.0f + __expf(-fabsf(t0 - t1)));
    const bool low = conf < 0.8f;
    out[r * 2 + 0] = low ? f0 : t0;
    out[r * 2 + 1] = low ? f1 : t1;
}

extern "C" void kernel_launch(void* const* d_in, const int* in_sizes, int n_in,
                              void* d_out, int out_size, void* d_ws, size_t ws_size,
                              hipStream_t stream) {
    (void)in_sizes; (void)n_in; (void)out_size; (void)ws_size;
    const float* x    = (const float*)d_in[0];
    const float* t_w1 = (const float*)d_in[1];
    const float* t_b1 = (const float*)d_in[2];
    const float* t_w2 = (const float*)d_in[3];
    const float* t_b2 = (const float*)d_in[4];
    const float* f_w1 = (const float*)d_in[5];
    const float* f_b1 = (const float*)d_in[6];
    const float* f_w2 = (const float*)d_in[7];
    const float* f_b2 = (const float*)d_in[8];
    float* acc = (float*)d_ws;                 // [2][8192][2] f32 = 128 KB
    float* out = (float*)d_out;

    hipMemsetAsync(acc, 0, (size_t)2 * BSZ * 2 * sizeof(float), stream);

    dim3 grid(HH / BN, BSZ / BM, 2);           // (64, 64, 2 experts)
    fused_expert_gemm<<<grid, 256, 0, stream>>>(x, t_w1, t_b1, t_w2,
                                                f_w1, f_b1, f_w2, acc);

    confidence_select<<<dim3(BSZ / 256), 256, 0, stream>>>(acc, t_b2, f_b2, out);
}
